// CausalSelfAttention_506806141196
// MI455X (gfx1250) — compile-verified, matrix operand order checked by reading, not run on hardware
//
#include <hip/hip_runtime.h>
#include <stdint.h>

// CDNA5 / gfx1250, wave32. Matrix math via v_wmma_f32_16x16x32_f16,
// tile feeds via global_load_async_to_lds_b128 (ASYNCcnt path).

typedef __attribute__((ext_vector_type(16))) _Float16 v16h;
typedef __attribute__((ext_vector_type(8)))  float    v8f;

#define HIDDEN 1280
#define NHEADS 20
#define HDIM   64
#define BATCH  2
#define SEQ    2048
#define H3     (3 * HIDDEN)
#define ROWS   (BATCH * SEQ)   // 4096

// ---------------------------------------------------------------------------
// Async global -> LDS copy, 16 bytes per lane. LDS address operand is the
// wave-relative LDS byte address == low 32 bits of the flat shared pointer
// (ISA: LDS aperture maps flat addr[31:0] to the wave's LDS space).
// ---------------------------------------------------------------------------
static __device__ __forceinline__ void async_copy_b128(void* lds_dst, const void* gsrc) {
  const uint32_t lds = (uint32_t)(uintptr_t)lds_dst;
  asm volatile("global_load_async_to_lds_b128 %0, %1, off"
               :: "v"(lds), "v"(gsrc) : "memory");
}
static __device__ __forceinline__ void async_wait0() {
  asm volatile("s_wait_asynccnt 0" ::: "memory");
}

// ---------------------------------------------------------------------------
// Load a 16x32 f16 fragment (A-operand layout per CDNA5 ISA 7.12.2) from LDS.
// Lanes 0-15: row = lane, VGPR v<4 -> K = 2v,2v+1 ; v>=4 -> K = 16+2(v-4),...
// Lanes 16-31: same rows, K offset +8. Contiguous word pairs -> ds_load_b128.
// For B operands the tile is stored transposed ([n][k]) so the same gather
// produces the B-fragment layout.
// ---------------------------------------------------------------------------
static __device__ __forceinline__ v16h ld_frag16x32(const _Float16* base, int lane, int ld) {
  const int m  = lane & 15;
  const int kh = (lane >> 4) << 3;            // 0 or 8
  const uint32_t* row = (const uint32_t*)(base + m * ld);  // ld even -> 4B aligned
  union { v16h h; uint32_t u[8]; } f;
#pragma unroll
  for (int v = 0; v < 8; ++v) {
    const int k = ((v < 4) ? (2 * v) : (16 + 2 * (v - 4))) + kh;
    f.u[v] = row[k >> 1];
  }
  return f.h;
}

static __device__ __forceinline__ v8f wmma_f16(v16h a, v16h b, v8f c) {
  // (neg_a, A, neg_b, B, c_mod, C, reuse_a, reuse_b)
  return __builtin_amdgcn_wmma_f32_16x16x32_f16(false, a, false, b, (short)0, c, false, false);
}

// ---------------------------------------------------------------------------
// f32 -> f16 streaming convert
// ---------------------------------------------------------------------------
__global__ void k_cvt(const float* __restrict__ in, _Float16* __restrict__ out, int n) {
  const int i = blockIdx.x * 256 + threadIdx.x;
  if (i < n) out[i] = (_Float16)in[i];
}

// ---------------------------------------------------------------------------
// QKV GEMM: [4096,1280]f16 x [1280,3840]f16 + bias -> head-major f16 Q/K/V
// Block: 128 threads (4 waves), 64x64 C tile, K-step 32.
// ---------------------------------------------------------------------------
__global__ __launch_bounds__(128) void k_gemm_qkv(
    const _Float16* __restrict__ A, const _Float16* __restrict__ Bw,
    const float* __restrict__ bias,
    _Float16* __restrict__ qb, _Float16* __restrict__ kb, _Float16* __restrict__ vb) {
  __shared__ __align__(16) _Float16 As[64][40];   // [m][k]
  __shared__ __align__(16) _Float16 Bs[64][40];   // [n][k] (transposed)
  const int tid = threadIdx.x, lane = tid & 31, wave = tid >> 5;
  const int tm = blockIdx.x * 64, tn = blockIdx.y * 64;
  v8f acc[4] = {};

  for (int k0 = 0; k0 < HIDDEN; k0 += 32) {
    {   // A tile: 64 rows x 32 halfs via async global->LDS (no VGPR staging)
      const int r = tid >> 1, c = (tid & 1) << 4;
      const _Float16* src = A + (size_t)(tm + r) * HIDDEN + k0 + c;
      async_copy_b128(&As[r][c],     src);
      async_copy_b128(&As[r][c + 8], src + 8);
    }
    {   // B tile: rows k0..k0+31, cols tn..tn+63, scatter-transpose into LDS
      const int kr = tid >> 2, cc = (tid & 3) << 4;
      const _Float16* src = Bw + (size_t)(k0 + kr) * H3 + tn + cc;
#pragma unroll
      for (int i = 0; i < 16; ++i) Bs[cc + i][kr] = src[i];
      if (k0 + 32 < HIDDEN)                       // prefetch next weight tile
        __builtin_prefetch(src + (size_t)32 * H3, 0, 0);
    }
    async_wait0();
    __syncthreads();
    const v16h a = ld_frag16x32(&As[wave * 16][0], lane, 40);
#pragma unroll
    for (int t = 0; t < 4; ++t) {
      const v16h b = ld_frag16x32(&Bs[t * 16][0], lane, 40);
      acc[t] = wmma_f16(a, b, acc[t]);
    }
    __syncthreads();
  }

  // Epilogue: +bias, convert f16, scatter into [b, h, s, d] (fused transpose)
  const int hi = lane >> 4, nl = lane & 15;
#pragma unroll
  for (int t = 0; t < 4; ++t) {
    const int n   = tn + t * 16 + nl;
    const float bv = bias[n];
    const int sec = n / HIDDEN, rem = n % HIDDEN;
    const int h = rem >> 6, d = rem & 63;
    _Float16* dst = (sec == 0) ? qb : (sec == 1) ? kb : vb;
#pragma unroll
    for (int r = 0; r < 8; ++r) {
      const int m = tm + wave * 16 + hi * 8 + r;
      const int bi = m >> 11, s = m & 2047;
      dst[((size_t)((bi * NHEADS + h) * SEQ + s)) * HDIM + d] = (_Float16)(acc[t][r] + bv);
    }
  }
}

// ---------------------------------------------------------------------------
// Flash attention: one block per (b*h, q-tile of 64). 128 threads (4 waves),
// each wave owns 16 q-rows. Online softmax, f32 accumulators.
// ---------------------------------------------------------------------------
__global__ __launch_bounds__(128) void k_attn(
    const _Float16* __restrict__ Q, const _Float16* __restrict__ K,
    const _Float16* __restrict__ V, _Float16* __restrict__ ctx) {
  __shared__ __align__(16) _Float16 Qs[64][72];   // [q][hd]
  __shared__ __align__(16) _Float16 Ks[64][72];   // [kv][hd]  (= B-frag [n][k])
  __shared__ __align__(16) _Float16 Vs[64][72];   // [hd][kv]  (transposed)
  __shared__ __align__(16) _Float16 Ps[64][72];   // per-wave P rows
  const int tid = threadIdx.x, lane = tid & 31, wave = tid >> 5;
  const int qt = blockIdx.x;                      // q tile 0..31
  const int bh = blockIdx.y;                      // 0..39
  const int bidx = bh / NHEADS, h = bh % NHEADS;
  const size_t head_base = (size_t)bh * SEQ * HDIM;
  const int hi = lane >> 4, nl = lane & 15;

  {   // Q tile via async copies (contiguous in both global and LDS)
    const int r = tid >> 1, c = (tid & 1) << 5;
    const _Float16* src = Q + head_base + (size_t)(qt * 64 + r) * HDIM + c;
    async_copy_b128(&Qs[r][c],      src);
    async_copy_b128(&Qs[r][c + 8],  src + 8);
    async_copy_b128(&Qs[r][c + 16], src + 16);
    async_copy_b128(&Qs[r][c + 24], src + 24);
  }

  float m_i[8], l_i[8];
#pragma unroll
  for (int r = 0; r < 8; ++r) { m_i[r] = -INFINITY; l_i[r] = 0.f; }
  v8f acc[4] = {};

  for (int j = 0; j <= qt; ++j) {
    {   // K tile, row-major, async
      const int r = tid >> 1, c = (tid & 1) << 5;
      const _Float16* src = K + head_base + (size_t)(j * 64 + r) * HDIM + c;
      async_copy_b128(&Ks[r][c],      src);
      async_copy_b128(&Ks[r][c + 8],  src + 8);
      async_copy_b128(&Ks[r][c + 16], src + 16);
      async_copy_b128(&Ks[r][c + 24], src + 24);
    }
    {   // V tile, transposed into [hd][kv] (needs 16-bit scatter -> VGPR path)
      const int kr = tid >> 1, c0 = (tid & 1) << 5;
      const _Float16* src = V + head_base + (size_t)(j * 64 + kr) * HDIM + c0;
      _Float16 tmp[32];
      *(uint4*)&tmp[0]  = *(const uint4*)(src);
      *(uint4*)&tmp[8]  = *(const uint4*)(src + 8);
      *(uint4*)&tmp[16] = *(const uint4*)(src + 16);
      *(uint4*)&tmp[24] = *(const uint4*)(src + 24);
#pragma unroll
      for (int i = 0; i < 32; ++i) Vs[c0 + i][kr] = tmp[i];
    }
    async_wait0();
    __syncthreads();

    // S = Q * K^T   (contraction over hd=64, two K-steps of 32)
    v8f st[4];
    {
      const v16h aq0 = ld_frag16x32(&Qs[wave * 16][0], lane, 72);
      const v16h aq1 = ld_frag16x32(&Qs[wave * 16][32], lane, 72);
#pragma unroll
      for (int t = 0; t < 4; ++t) {
        v8f sacc = {};
        sacc = wmma_f16(aq0, ld_frag16x32(&Ks[t * 16][0], lane, 72), sacc);
        sacc = wmma_f16(aq1, ld_frag16x32(&Ks[t * 16][32], lane, 72), sacc);
        st[t] = sacc;
      }
    }

    // scale + causal mask (only diagonal tile needs masking)
    const float scl = 0.125f;                     // 1/sqrt(64)
    const bool diag = (j == qt);
    float sv[4][8];
#pragma unroll
    for (int t = 0; t < 4; ++t) {
      const int kv = j * 64 + t * 16 + nl;
#pragma unroll
      for (int r = 0; r < 8; ++r) {
        const int qrow = qt * 64 + wave * 16 + hi * 8 + r;
        float x = st[t][r] * scl;
        if (diag && kv > qrow) x = -INFINITY;
        sv[t][r] = x;
      }
    }

    // online softmax: row reductions across the 16 lanes holding each row
    float mnew[8], alpha[8], rs[8];
#pragma unroll
    for (int r = 0; r < 8; ++r) {
      float mx = fmaxf(fmaxf(sv[0][r], sv[1][r]), fmaxf(sv[2][r], sv[3][r]));
#pragma unroll
      for (int o = 8; o >= 1; o >>= 1) mx = fmaxf(mx, __shfl_xor(mx, o, 32));
      mnew[r]  = fmaxf(m_i[r], mx);
      alpha[r] = __expf(m_i[r] - mnew[r]);
      m_i[r]   = mnew[r];
      rs[r]    = 0.f;
    }
#pragma unroll
    for (int t = 0; t < 4; ++t) {
#pragma unroll
      for (int r = 0; r < 8; ++r) {
        const float p = __expf(sv[t][r] - mnew[r]);
        rs[r] += p;
        Ps[wave * 16 + hi * 8 + r][t * 16 + nl] = (_Float16)p;  // re-layout via LDS
      }
    }
#pragma unroll
    for (int r = 0; r < 8; ++r) {
      float ssum = rs[r];
#pragma unroll
      for (int o = 8; o >= 1; o >>= 1) ssum += __shfl_xor(ssum, o, 32);
      l_i[r] = l_i[r] * alpha[r] + ssum;
#pragma unroll
      for (int t = 0; t < 4; ++t) acc[t][r] *= alpha[r];
    }

    // acc += P * V   (contraction over kv=64, two K-steps of 32)
    {
      const v16h ap0 = ld_frag16x32(&Ps[wave * 16][0], lane, 72);
      const v16h ap1 = ld_frag16x32(&Ps[wave * 16][32], lane, 72);
#pragma unroll
      for (int t = 0; t < 4; ++t) {
        acc[t] = wmma_f16(ap0, ld_frag16x32(&Vs[t * 16][0], lane, 72), acc[t]);
        acc[t] = wmma_f16(ap1, ld_frag16x32(&Vs[t * 16][32], lane, 72), acc[t]);
      }
    }
    __syncthreads();
  }

  // out = acc / l, un-transpose heads: ctx[b, s, h*64+d]
#pragma unroll
  for (int t = 0; t < 4; ++t) {
#pragma unroll
    for (int r = 0; r < 8; ++r) {
      const int srow = qt * 64 + wave * 16 + hi * 8 + r;
      const int d = t * 16 + nl;
      ctx[((size_t)(bidx * SEQ + srow)) * HIDDEN + h * HDIM + d] =
          (_Float16)(acc[t][r] / l_i[r]);
    }
  }
}

// ---------------------------------------------------------------------------
// Output projection: [4096,1280]f16 x [1280,1280]f16 + bias -> f32 out
// ---------------------------------------------------------------------------
__global__ __launch_bounds__(128) void k_gemm_proj(
    const _Float16* __restrict__ A, const _Float16* __restrict__ Bw,
    const float* __restrict__ bias, float* __restrict__ out) {
  __shared__ __align__(16) _Float16 As[64][40];
  __shared__ __align__(16) _Float16 Bs[64][40];
  const int tid = threadIdx.x, lane = tid & 31, wave = tid >> 5;
  const int tm = blockIdx.x * 64, tn = blockIdx.y * 64;
  v8f acc[4] = {};

  for (int k0 = 0; k0 < HIDDEN; k0 += 32) {
    {
      const int r = tid >> 1, c = (tid & 1) << 4;
      const _Float16* src = A + (size_t)(tm + r) * HIDDEN + k0 + c;
      async_copy_b128(&As[r][c],     src);
      async_copy_b128(&As[r][c + 8], src + 8);
    }
    {
      const int kr = tid >> 2, cc = (tid & 3) << 4;
      const _Float16* src = Bw + (size_t)(k0 + kr) * HIDDEN + tn + cc;
#pragma unroll
      for (int i = 0; i < 16; ++i) Bs[cc + i][kr] = src[i];
      if (k0 + 32 < HIDDEN)
        __builtin_prefetch(src + (size_t)32 * HIDDEN, 0, 0);
    }
    async_wait0();
    __syncthreads();
    const v16h a = ld_frag16x32(&As[wave * 16][0], lane, 40);
#pragma unroll
    for (int t = 0; t < 4; ++t) {
      const v16h b = ld_frag16x32(&Bs[t * 16][0], lane, 40);
      acc[t] = wmma_f16(a, b, acc[t]);
    }
    __syncthreads();
  }

  const int hi = lane >> 4, nl = lane & 15;
#pragma unroll
  for (int t = 0; t < 4; ++t) {
    const int n = tn + t * 16 + nl;
    const float bv = bias[n];
#pragma unroll
    for (int r = 0; r < 8; ++r) {
      const int m = tm + wave * 16 + hi * 8 + r;
      out[(size_t)m * HIDDEN + n] = acc[t][r] + bv;
    }
  }
}

// ---------------------------------------------------------------------------
// Host launch
// ---------------------------------------------------------------------------
extern "C" void kernel_launch(void* const* d_in, const int* in_sizes, int n_in,
                              void* d_out, int out_size, void* d_ws, size_t ws_size,
                              hipStream_t stream) {
  const float* x      = (const float*)d_in[0];   // [2,2048,1280]
  const float* w_attn = (const float*)d_in[1];   // [1280,3840]
  const float* b_attn = (const float*)d_in[2];   // [3840]
  const float* w_proj = (const float*)d_in[3];   // [1280,1280]
  const float* b_proj = (const float*)d_in[4];   // [1280]
  float* out = (float*)d_out;                    // [2,2048,1280] f32

  char* ws = (char*)d_ws;
  size_t off = 0;
  auto balloc = [&](size_t bytes) -> void* {
    void* p = ws + off;
    off = (off + bytes + 255) & ~(size_t)255;
    return p;
  };
  _Float16* xh  = (_Float16*)balloc((size_t)ROWS * HIDDEN * 2);
  _Float16* wah = (_Float16*)balloc((size_t)HIDDEN * H3 * 2);
  _Float16* wph = (_Float16*)balloc((size_t)HIDDEN * HIDDEN * 2);
  _Float16* qb  = (_Float16*)balloc((size_t)ROWS * HIDDEN * 2);
  _Float16* kb  = (_Float16*)balloc((size_t)ROWS * HIDDEN * 2);
  _Float16* vb  = (_Float16*)balloc((size_t)ROWS * HIDDEN * 2);
  _Float16* ctx = (_Float16*)balloc((size_t)ROWS * HIDDEN * 2);
  (void)ws_size; (void)in_sizes; (void)n_in; (void)out_size;

  k_cvt<<<(ROWS * HIDDEN + 255) / 256, 256, 0, stream>>>(x, xh, ROWS * HIDDEN);
  k_cvt<<<(HIDDEN * H3 + 255) / 256, 256, 0, stream>>>(w_attn, wah, HIDDEN * H3);
  k_cvt<<<(HIDDEN * HIDDEN + 255) / 256, 256, 0, stream>>>(w_proj, wph, HIDDEN * HIDDEN);

  k_gemm_qkv<<<dim3(ROWS / 64, H3 / 64), 128, 0, stream>>>(xh, wah, b_attn, qb, kb, vb);
  k_attn<<<dim3(SEQ / 64, BATCH * NHEADS), 128, 0, stream>>>(qb, kb, vb, ctx);
  k_gemm_proj<<<dim3(ROWS / 64, HIDDEN / 64), 128, 0, stream>>>(ctx, wph, b_proj, out);
}
